// MoE_59519656788313
// MI455X (gfx1250) — matrix operand order repaired and checked by reading, not compiled, not run on hardware
//
#include <hip/hip_runtime.h>

// ---------------------------------------------------------------------------
// MoE (top-2 of 8 experts), D=1024, H=2048, N=8192 tokens.
// Compute-bound (AI ~800 FLOP/B) => bf16 WMMA (v_wmma_f32_16x16x32_bf16).
// Weights are converted f32->bf16 AND transposed to [E][N][K] once per call so
// all GEMM tiles stage with vectorized async global->LDS (b128) copies.
// Double-buffered LDS + ASYNCcnt pipeline overlaps DMA with WMMA.
// Workspace usage: ~252 MB.
// ---------------------------------------------------------------------------

#define N_TOK 8192
#define DIM   1024
#define HID   2048
#define H2    4096
#define NEXP  8
#define TOPK  2

typedef __attribute__((ext_vector_type(16))) __bf16 v16bf;
typedef __attribute__((ext_vector_type(8)))  __bf16 v8bf;
typedef __attribute__((ext_vector_type(8)))  float  v8f;

union FragA { v16bf v; v8bf h[2]; };

// f32 -> bf16 round-to-nearest-even
static __device__ __forceinline__ unsigned short f2bf(float f) {
    unsigned int u = __float_as_uint(f);
    u += 0x7FFFu + ((u >> 16) & 1u);
    return (unsigned short)(u >> 16);
}

// Async global->LDS copy, 16B per lane, tracked by ASYNCcnt.
// lds_off: LDS byte address (low 32 bits of generic shared pointer).
// voff:    per-lane signed byte offset from 64-bit SGPR base.
static __device__ __forceinline__ void async_b128(unsigned int lds_off, int voff,
                                                  const void* base) {
    asm volatile("global_load_async_to_lds_b128 %0, %1, %2"
                 :: "v"(lds_off), "v"(voff), "s"(base) : "memory");
}
static __device__ __forceinline__ unsigned int lds_addr(const void* p) {
    return (unsigned int)(unsigned long long)p;
}

// ---------------------------------------------------------------------------
// Kernel 0a: f32 -> bf16 bulk convert (x stays row-major [N][D])
// ---------------------------------------------------------------------------
__global__ __launch_bounds__(256)
void cvt_f32_bf16(const float* __restrict__ src, unsigned short* __restrict__ dst, int n4) {
    int i = blockIdx.x * 256 + threadIdx.x;
    if (i >= n4) return;
    float4 v = reinterpret_cast<const float4*>(src)[i];
    union { uint2 q; unsigned short u[4]; } o;
    o.u[0] = f2bf(v.x); o.u[1] = f2bf(v.y); o.u[2] = f2bf(v.z); o.u[3] = f2bf(v.w);
    reinterpret_cast<uint2*>(dst)[i] = o.q;
}

// ---------------------------------------------------------------------------
// Kernel 0b: f32 [E][K][N] -> bf16 [E][N][K] transpose (32x32 LDS tiles).
// Makes every GEMM operand "contiguous along K" => vector async LDS staging.
// ---------------------------------------------------------------------------
__global__ __launch_bounds__(256)
void cvt_transpose_bf16(const float* __restrict__ src, unsigned short* __restrict__ dst,
                        int K, int N) {
    __shared__ float s[32][33];
    const int e  = blockIdx.z;
    const int n0 = blockIdx.x * 32, k0 = blockIdx.y * 32;
    const int tid = threadIdx.x;
    const int r = tid >> 5, c = tid & 31;                 // 8 rows x 32 cols
    const float* sp = src + (size_t)e * K * N;
#pragma unroll
    for (int i = 0; i < 4; ++i)
        s[r + i * 8][c] = sp[(size_t)(k0 + r + i * 8) * N + (n0 + c)];
    __syncthreads();
    unsigned short* dp = dst + (size_t)e * K * N;
#pragma unroll
    for (int i = 0; i < 4; ++i)
        dp[(size_t)(n0 + r + i * 8) * K + (k0 + c)] = f2bf(s[c][r + i * 8]);
}

// ---------------------------------------------------------------------------
// Kernel 1: router. One wave32 per token: logits = x[n]·Wr + br, top-2,
// softmax over the two selected logits, bucket the assignment per expert.
// assignRow = n*2 + k. Bucket order is atomic-nondeterministic but each
// assignRow's math is independent => output deterministic.
// ---------------------------------------------------------------------------
__global__ __launch_bounds__(256)
void router_kernel(const float* __restrict__ x, const float* __restrict__ Wr,
                   const float* __restrict__ br, int* __restrict__ counts,
                   int* __restrict__ exlist, float* __restrict__ score) {
    const int warp = threadIdx.x >> 5;
    const int lane = threadIdx.x & 31;
    const int n = blockIdx.x * 8 + warp;
    if (n >= N_TOK) return;

    float acc[NEXP];
#pragma unroll
    for (int e = 0; e < NEXP; ++e) acc[e] = 0.f;

    const float* xr = x + (size_t)n * DIM;
    for (int d = lane; d < DIM; d += 32) {
        float xv = xr[d];
        const float* wr = Wr + (size_t)d * NEXP;
#pragma unroll
        for (int e = 0; e < NEXP; ++e) acc[e] += xv * wr[e];
    }
#pragma unroll
    for (int e = 0; e < NEXP; ++e) {
#pragma unroll
        for (int off = 16; off > 0; off >>= 1)
            acc[e] += __shfl_xor(acc[e], off, 32);
    }

    if (lane == 0) {
        float lg[NEXP];
#pragma unroll
        for (int e = 0; e < NEXP; ++e) lg[e] = acc[e] + br[e];
        int i0 = 0; float v0 = lg[0];
#pragma unroll
        for (int e = 1; e < NEXP; ++e) if (lg[e] > v0) { v0 = lg[e]; i0 = e; }
        int i1 = -1; float v1 = -3.4e38f;
#pragma unroll
        for (int e = 0; e < NEXP; ++e)
            if (e != i0 && lg[e] > v1) { v1 = lg[e]; i1 = e; }
        float e1  = __expf(v1 - v0);
        float inv = 1.f / (1.f + e1);
        float s0 = inv, s1 = e1 * inv;

        int p0 = atomicAdd(&counts[i0], 1);
        exlist[i0 * N_TOK + p0] = n * TOPK + 0;
        score[n * TOPK + 0] = s0;
        int p1 = atomicAdd(&counts[i1], 1);
        exlist[i1 * N_TOK + p1] = n * TOPK + 1;
        score[n * TOPK + 1] = s1;
    }
}

// ---------------------------------------------------------------------------
// Kernel 2: grouped GEMM1 + SwiGLU.
// Block tile: 128 gathered tokens x 64 hidden cols; both a-half and g-half of
// W1T computed by the block so act = silu(a)*g fuses in registers.
// 8 waves: 4(M) x 2(N); each wave 32x32 per half => 8 WMMAs / k-step.
// Double-buffered LDS fed by async global->LDS (4 ops/thread/tile).
// ---------------------------------------------------------------------------
__global__ __launch_bounds__(256)
void moe_gemm1_swiglu(const unsigned short* __restrict__ xb,
                      const unsigned short* __restrict__ W1T,   // [E][4096][1024] bf16
                      const float* __restrict__ b1,
                      const int* __restrict__ counts,
                      const int* __restrict__ exlist,
                      unsigned short* __restrict__ act) {
    const int e  = blockIdx.z;
    const int jt = blockIdx.x;   // 64-col slab of H (0..31)
    const int mt = blockIdx.y;   // 128-token tile (0..63)
    const int cnt = counts[e];
    if (mt * 128 >= cnt) return;

    __shared__ int sAssign[128];
    __shared__ __align__(16) unsigned short sA[2][128 * 32];
    __shared__ __align__(32) unsigned short sB[2][2][64 * 32];   // [buf][a|g][n][k]

    const int tid = threadIdx.x;
    if (tid < 128) {
        int idx = mt * 128 + tid;
        sAssign[tid] = (idx < cnt) ? exlist[e * N_TOK + idx] : -1;
    }
    __syncthreads();

    // A loader: 32B/thread  (row, 16-elem half)
    const int rowA = tid >> 1, halfA = tid & 1;
    int ta = sAssign[rowA];
    const int tokA = (ta < 0) ? 0 : (ta >> 1);
    // B loader: 16B/thread/half  (n row, 8-elem quarter)
    const int nB = tid >> 2, kqB = tid & 3;

    const int wave = tid >> 5, lane = tid & 31;
    const int wm = wave >> 1, wn = wave & 1;
    const int lm = lane & 15, lh = lane >> 4;

    const v8f vz = {0.f,0.f,0.f,0.f,0.f,0.f,0.f,0.f};
    v8f accA[2][2], accG[2][2];
#pragma unroll
    for (int mf = 0; mf < 2; ++mf)
#pragma unroll
        for (int nf = 0; nf < 2; ++nf) { accA[mf][nf] = vz; accG[mf][nf] = vz; }

    const int colAa = e * H2 + jt * 64;          // row in W1T (a-half)
    const int colAg = e * H2 + HID + jt * 64;    // row in W1T (g-half)

    const unsigned int ldsA0 = lds_addr(&sA[0][rowA * 32 + halfA * 16]);
    const unsigned int ldsBa0 = lds_addr(&sB[0][0][nB * 32 + kqB * 8]);
    const unsigned int ldsBg0 = lds_addr(&sB[0][1][nB * 32 + kqB * 8]);
    const unsigned int bufStrA = (unsigned int)(128 * 32 * 2);
    const unsigned int bufStrB = (unsigned int)(2 * 64 * 32 * 2);

    const int vaBase  = (tokA * DIM + halfA * 16) * 2;
    const int vbaBase = ((colAa + nB) * DIM + kqB * 8) * 2;
    const int vbgBase = ((colAg + nB) * DIM + kqB * 8) * 2;

#define G1_PREFETCH(buf, k0)                                                    \
    do {                                                                        \
        int kb = (k0) * 2;                                                      \
        async_b128(ldsA0 + (buf) * bufStrA,       vaBase + kb,       xb);       \
        async_b128(ldsA0 + (buf) * bufStrA + 16,  vaBase + kb + 16,  xb);       \
        async_b128(ldsBa0 + (buf) * bufStrB,      vbaBase + kb,      W1T);      \
        async_b128(ldsBg0 + (buf) * bufStrB,      vbgBase + kb,      W1T);      \
    } while (0)

    G1_PREFETCH(0, 0);
    int cur = 0;
    for (int kt = 0; kt < DIM / 32; ++kt) {
        if (kt + 1 < DIM / 32) {
            G1_PREFETCH(cur ^ 1, (kt + 1) * 32);
            asm volatile("s_wait_asynccnt 0x4" ::: "memory");  // prev tile landed
        } else {
            asm volatile("s_wait_asynccnt 0x0" ::: "memory");
        }
        __syncthreads();

        FragA afr[2];
#pragma unroll
        for (int mf = 0; mf < 2; ++mf) {
            int r = wm * 32 + mf * 16 + lm;
            afr[mf].h[0] = *reinterpret_cast<const v8bf*>(&sA[cur][r * 32 + lh * 16]);
            afr[mf].h[1] = *reinterpret_cast<const v8bf*>(&sA[cur][r * 32 + lh * 16 + 8]);
        }
        v16bf bfa[2], bfg[2];
#pragma unroll
        for (int nf = 0; nf < 2; ++nf) {
            int c = wn * 32 + nf * 16 + lm;
            bfa[nf] = *reinterpret_cast<const v16bf*>(&sB[cur][0][c * 32 + lh * 16]);
            bfg[nf] = *reinterpret_cast<const v16bf*>(&sB[cur][1][c * 32 + lh * 16]);
        }
#pragma unroll
        for (int mf = 0; mf < 2; ++mf)
#pragma unroll
            for (int nf = 0; nf < 2; ++nf) {
                accA[mf][nf] = __builtin_amdgcn_wmma_f32_16x16x32_bf16(
                    false, afr[mf].v, false, bfa[nf], (short)0, accA[mf][nf], false, false);
                accG[mf][nf] = __builtin_amdgcn_wmma_f32_16x16x32_bf16(
                    false, afr[mf].v, false, bfg[nf], (short)0, accG[mf][nf], false, false);
            }
        __syncthreads();   // all reads done before next prefetch overwrites buf
        cur ^= 1;
    }
#undef G1_PREFETCH

    // epilogue: bias, SwiGLU, store bf16 act rows (indexed by assignRow)
#pragma unroll
    for (int mf = 0; mf < 2; ++mf)
#pragma unroll
        for (int nf = 0; nf < 2; ++nf) {
            int c = jt * 64 + wn * 32 + nf * 16 + lm;    // [0, 2048)
            float ba = b1[(size_t)e * H2 + c];
            float bg = b1[(size_t)e * H2 + HID + c];
#pragma unroll
            for (int v = 0; v < 8; ++v) {
                int r = wm * 32 + mf * 16 + lh * 8 + v;
                int as = sAssign[r];
                if (as >= 0) {
                    float a = accA[mf][nf][v] + ba;
                    float g = accG[mf][nf][v] + bg;
                    float s = a * __builtin_amdgcn_rcpf(1.f + __expf(-a));  // silu
                    act[(size_t)as * HID + c] = f2bf(s * g);
                }
            }
        }
}

// ---------------------------------------------------------------------------
// Kernel 3: grouped GEMM2: out_row[assign] = (act[assign] @ W2 + b2)*score
// Block tile: 128 rows x 64 D cols; K over H=2048. Async double-buffered.
// ---------------------------------------------------------------------------
__global__ __launch_bounds__(256)
void moe_gemm2(const unsigned short* __restrict__ act,
               const unsigned short* __restrict__ W2T,   // [E][1024][2048] bf16
               const float* __restrict__ b2,
               const float* __restrict__ score,
               const int* __restrict__ counts,
               const int* __restrict__ exlist,
               float* __restrict__ orows) {
    const int e  = blockIdx.z;
    const int dt = blockIdx.x;   // 64-col slab of D (0..15)
    const int mt = blockIdx.y;
    const int cnt = counts[e];
    if (mt * 128 >= cnt) return;

    __shared__ int   sAssign[128];
    __shared__ float sSc[128];
    __shared__ __align__(16) unsigned short sA[2][128 * 32];
    __shared__ __align__(32) unsigned short sB[2][64 * 32];

    const int tid = threadIdx.x;
    if (tid < 128) {
        int idx = mt * 128 + tid;
        int as = (idx < cnt) ? exlist[e * N_TOK + idx] : -1;
        sAssign[tid] = as;
        sSc[tid] = (as >= 0) ? score[as] : 0.f;
    }
    __syncthreads();

    const int rowA = tid >> 1, halfA = tid & 1;
    int aa = sAssign[rowA];
    const int aRowIdx = (aa < 0) ? 0 : aa;
    const int nB = tid >> 2, kqB = tid & 3;

    const int wave = tid >> 5, lane = tid & 31;
    const int wm = wave >> 1, wn = wave & 1;
    const int lm = lane & 15, lh = lane >> 4;

    const v8f vz = {0.f,0.f,0.f,0.f,0.f,0.f,0.f,0.f};
    v8f acc[2][2];
#pragma unroll
    for (int mf = 0; mf < 2; ++mf)
#pragma unroll
        for (int nf = 0; nf < 2; ++nf) acc[mf][nf] = vz;

    const int colB = e * DIM + dt * 64;

    const unsigned int ldsA0 = lds_addr(&sA[0][rowA * 32 + halfA * 16]);
    const unsigned int ldsB0 = lds_addr(&sB[0][nB * 32 + kqB * 8]);
    const unsigned int bufStrA = (unsigned int)(128 * 32 * 2);
    const unsigned int bufStrB = (unsigned int)(64 * 32 * 2);

    const int vaBase = (aRowIdx * HID + halfA * 16) * 2;
    const int vbBase = ((colB + nB) * HID + kqB * 8) * 2;

#define G2_PREFETCH(buf, k0)                                                   \
    do {                                                                       \
        int kb = (k0) * 2;                                                     \
        async_b128(ldsA0 + (buf) * bufStrA,      vaBase + kb,      act);       \
        async_b128(ldsA0 + (buf) * bufStrA + 16, vaBase + kb + 16, act);       \
        async_b128(ldsB0 + (buf) * bufStrB,      vbBase + kb,      W2T);       \
    } while (0)

    G2_PREFETCH(0, 0);
    int cur = 0;
    for (int kt = 0; kt < HID / 32; ++kt) {
        if (kt + 1 < HID / 32) {
            G2_PREFETCH(cur ^ 1, (kt + 1) * 32);
            asm volatile("s_wait_asynccnt 0x3" ::: "memory");
        } else {
            asm volatile("s_wait_asynccnt 0x0" ::: "memory");
        }
        __syncthreads();

        FragA afr[2];
#pragma unroll
        for (int mf = 0; mf < 2; ++mf) {
            int r = wm * 32 + mf * 16 + lm;
            afr[mf].h[0] = *reinterpret_cast<const v8bf*>(&sA[cur][r * 32 + lh * 16]);
            afr[mf].h[1] = *reinterpret_cast<const v8bf*>(&sA[cur][r * 32 + lh * 16 + 8]);
        }
        v16bf bfr[2];
#pragma unroll
        for (int nf = 0; nf < 2; ++nf) {
            int c = wn * 32 + nf * 16 + lm;
            bfr[nf] = *reinterpret_cast<const v16bf*>(&sB[cur][c * 32 + lh * 16]);
        }
#pragma unroll
        for (int mf = 0; mf < 2; ++mf)
#pragma unroll
            for (int nf = 0; nf < 2; ++nf)
                acc[mf][nf] = __builtin_amdgcn_wmma_f32_16x16x32_bf16(
                    false, afr[mf].v, false, bfr[nf], (short)0, acc[mf][nf], false, false);
        __syncthreads();
        cur ^= 1;
    }
#undef G2_PREFETCH

#pragma unroll
    for (int mf = 0; mf < 2; ++mf)
#pragma unroll
        for (int nf = 0; nf < 2; ++nf) {
            int c = dt * 64 + wn * 32 + nf * 16 + lm;    // [0, 1024)
            float bb = b2[(size_t)e * DIM + c];
#pragma unroll
            for (int v = 0; v < 8; ++v) {
                int r = wm * 32 + mf * 16 + lh * 8 + v;
                int as = sAssign[r];
                if (as >= 0)
                    orows[(size_t)as * DIM + c] = (acc[mf][nf][v] + bb) * sSc[r];
            }
        }
}

// ---------------------------------------------------------------------------
// Kernel 4: combine top-2 rows per token (fixed row pairing => deterministic)
// ---------------------------------------------------------------------------
__global__ __launch_bounds__(256)
void combine_kernel(const float* __restrict__ rows, float* __restrict__ out) {
    int i = blockIdx.x * 256 + threadIdx.x;       // over N*D/4
    if (i >= N_TOK * DIM / 4) return;
    int n  = i >> 8;                              // D/4 = 256 float4 per row
    int d4 = i & 255;
    const float4* r = reinterpret_cast<const float4*>(rows);
    float4 a = r[(size_t)(2 * n) * 256 + d4];
    float4 b = r[(size_t)(2 * n + 1) * 256 + d4];
    float4 o = {a.x + b.x, a.y + b.y, a.z + b.z, a.w + b.w};
    reinterpret_cast<float4*>(out)[i] = o;
}

// ---------------------------------------------------------------------------
// Workspace layout (bytes) — total ~252 MB
// ---------------------------------------------------------------------------
#define OFF_XB     ((size_t)0)                          // 16 MB  bf16 x [N][D]
#define OFF_W1B    ((size_t)16777216)                   // 64 MB  bf16 W1^T [E][4096][1024]
#define OFF_W2B    ((size_t)83886080)                   // 32 MB  bf16 W2^T [E][1024][2048]
#define OFF_ACT    ((size_t)117440512)                  // 64 MB  bf16 act [N*K][H]
#define OFF_OROWS  ((size_t)184549376)                  // 64 MB  f32 out rows [N*K][D]
#define OFF_CNT    ((size_t)251658240)                  // 32 B   counts
#define OFF_LIST   ((size_t)251658272)                  // 256 KB expert lists
#define OFF_SCORE  ((size_t)251920416)                  // 64 KB  scores

extern "C" void kernel_launch(void* const* d_in, const int* in_sizes, int n_in,
                              void* d_out, int out_size, void* d_ws, size_t ws_size,
                              hipStream_t stream) {
    const float* x  = (const float*)d_in[0];
    const float* W1 = (const float*)d_in[1];
    const float* b1 = (const float*)d_in[2];
    const float* W2 = (const float*)d_in[3];
    const float* b2 = (const float*)d_in[4];
    const float* Wr = (const float*)d_in[5];
    const float* br = (const float*)d_in[6];
    float* out = (float*)d_out;

    char* ws = (char*)d_ws;
    unsigned short* xb     = (unsigned short*)(ws + OFF_XB);
    unsigned short* W1T    = (unsigned short*)(ws + OFF_W1B);
    unsigned short* W2T    = (unsigned short*)(ws + OFF_W2B);
    unsigned short* act    = (unsigned short*)(ws + OFF_ACT);
    float*          orows  = (float*)(ws + OFF_OROWS);
    int*            counts = (int*)(ws + OFF_CNT);
    int*            exlist = (int*)(ws + OFF_LIST);
    float*          score  = (float*)(ws + OFF_SCORE);

    hipMemsetAsync(counts, 0, NEXP * sizeof(int), stream);

    cvt_f32_bf16<<<(N_TOK * DIM / 4 + 255) / 256, 256, 0, stream>>>(x, xb, N_TOK * DIM / 4);
    // W1 [E][1024][4096] -> W1T [E][4096][1024]
    cvt_transpose_bf16<<<dim3(H2 / 32, DIM / 32, NEXP), 256, 0, stream>>>(W1, W1T, DIM, H2);
    // W2 [E][2048][1024] -> W2T [E][1024][2048]
    cvt_transpose_bf16<<<dim3(DIM / 32, HID / 32, NEXP), 256, 0, stream>>>(W2, W2T, HID, DIM);

    router_kernel<<<N_TOK / 8, 256, 0, stream>>>(x, Wr, br, counts, exlist, score);

    moe_gemm1_swiglu<<<dim3(HID / 64, N_TOK / 128, NEXP), 256, 0, stream>>>(
        xb, W1T, b1, counts, exlist, act);

    moe_gemm2<<<dim3(DIM / 64, N_TOK / 128, NEXP), 256, 0, stream>>>(
        act, W2T, b2, score, counts, exlist, orows);

    combine_kernel<<<(N_TOK * DIM / 4 + 255) / 256, 256, 0, stream>>>(orows, out);
}
